// JTNNEncoder_71743133712716
// MI455X (gfx1250) — matrix operand order repaired and checked by reading, not compiled, not run on hardware
//
#include <hip/hip_runtime.h>

#define H 512
#define E 8192
#define MAX_NB 8
#define N_ITERS 16
#define VOCAB 1024
#define B_ROOT 8192
#define N_MSG (N_ITERS * E)

typedef __bf16 bf16;
typedef __attribute__((ext_vector_type(2)))  __bf16 v2bf;
typedef __attribute__((ext_vector_type(16))) __bf16 v16bf;
typedef __attribute__((ext_vector_type(8)))  float  v8f;

static __device__ __forceinline__ v8f wmma_bf16(v16bf a, v16bf b, v8f c) {
    // v_wmma_f32_16x16x32_bf16: D = A(16x32) * B(32x16) + C(16x16 f32)
    return __builtin_amdgcn_wmma_f32_16x16x32_bf16(false, a, false, b, (short)0, c, false, false);
}
static __device__ __forceinline__ float sigmoidf_(float x) { return 1.f / (1.f + __expf(-x)); }
static __device__ __forceinline__ float tanhf_(float x) {
    float e = __expf(2.f * x);
    return (e - 1.f) / (e + 1.f);
}
// Pack two floats into one dword of bf16x2 -> single v_cvt_pk_bf16_f32.
static __device__ __forceinline__ unsigned pack_bf16x2(float x, float y) {
    v2bf pv;
    pv.x = (bf16)x;
    pv.y = (bf16)y;
    return __builtin_bit_cast(unsigned, pv);
}

// --- A-fragment staging into LDS, pre-swizzled to the contiguous-k order used here for A and B ---
// A 16x32 bf16 (this code's k order, not AMD's table): lane<16: m=lane,   VGPR v holds K=2*v+{0,1}
//               lane>=16: m=lane-16, same K pattern + 16.
// Stored as lds[tile][lane][16] so each lane reads one contiguous v16bf.
// A K-pair (kk even, kk+1) lands in one dword -> single 32-bit LDS store.
static __device__ __forceinline__ void stA(bf16* lds, int mt, int m, int kk, float x, float y) {
    int hi   = (kk >> 4) & 1;
    int lane = m + (hi << 4);
    int v    = (kk & 15) >> 1;
    bf16* p  = lds + (((mt << 5) + lane) << 4) + (v << 1);   // 4B aligned
    *(unsigned*)p = pack_bf16x2(x, y);
}

// ---------------------------------------------------------------------------
// Pack a [512,512] fp32 weight into bf16 transposed BT[n][k]. A WMMA B-frag
// is then 32 contiguous bytes per lane: BT[n*512 + ks*32 + hi*16 .. +15].
// One thread packs a K-pair -> single 32-bit global store.
// ---------------------------------------------------------------------------
__global__ void k_pack_bt(const float* __restrict__ W, bf16* __restrict__ BT) {
    int idx = blockIdx.x * blockDim.x + threadIdx.x;  // H*H/2 threads
    int n = idx >> 8, k2 = (idx & 255) << 1;
    float x = W[(size_t)k2 * H + n];
    float y = W[(size_t)(k2 + 1) * H + n];
    *(unsigned*)(BT + (size_t)n * H + k2) = pack_bf16x2(x, y);
}

// ---------------------------------------------------------------------------
// Root-only gather-sum: out[e] = sum_j h[nei[e][j]]
// ---------------------------------------------------------------------------
__global__ void k_gather_sum(const float* __restrict__ h, const int* __restrict__ nei,
                             float* __restrict__ out, int nrows) {
    int idx = blockIdx.x * blockDim.x + threadIdx.x;
    if (idx >= nrows * 128) return;
    int e = idx >> 7, c4 = (idx & 127) << 2;
    float4 s = {0.f, 0.f, 0.f, 0.f};
#pragma unroll
    for (int j = 0; j < MAX_NB; j++) {
        int row = nei[(e << 3) + j];
        const float4 v = *(const float4*)(h + (size_t)row * H + c4);
        s.x += v.x; s.y += v.y; s.z += v.z; s.w += v.w;
    }
    *(float4*)(out + (size_t)e * H + c4) = s;
}

// ---------------------------------------------------------------------------
// r2 gather-GEMM + fused epilogue:
//   r      = sigmoid(embR[wid] + h_nei @ U_r + b)
//   sum_gh = sum_j r * h_nei      sum_h = sum_j h_nei   (both in-lane)
// Block: 16 edges -> 128 gathered A rows; N coverage 128 (8 waves x 1 tile).
// A is double-buffered in LDS (one barrier/k-step); row pointers are hoisted
// so per-step staging is pure global_load_b64 with immediate offsets; B frags
// load directly from packed BT (L2-resident).
// ---------------------------------------------------------------------------
__global__ void __launch_bounds__(256)
k_r2_gemm(const float* __restrict__ h, const int* __restrict__ nei,
          const int* __restrict__ wid, const float* __restrict__ embR,
          const bf16* __restrict__ BTUr, const float* __restrict__ Urb,
          float* __restrict__ sum_h, float* __restrict__ sum_gh) {
    __shared__ __align__(32) bf16 ldsA[2][8 * 32 * 16];  // 2 x (128 rows x 32 K)
    __shared__ int s_rows[128];
    __shared__ int s_wid[16];

    const int t  = threadIdx.x;
    const int e0 = blockIdx.x * 16;
    const int n0 = blockIdx.y * 128;
    if (t < 128)      s_rows[t]      = nei[(size_t)e0 * MAX_NB + t];
    else if (t < 144) s_wid[t - 128] = wid[e0 + (t - 128)];
    __syncthreads();

    const int wave = t >> 5, lane = t & 31;
    const int hi = lane >> 4, nn = lane & 15;
    v8f acc[8];
#pragma unroll
    for (int mt = 0; mt < 8; mt++)
#pragma unroll
        for (int i = 0; i < 8; i++) acc[mt][i] = 0.f;

    // hoist loop-invariant staging state: row pointers + LDS slot addresses
    const float* rowp[8];
    int smt[8], sm[8], skp[8];
#pragma unroll
    for (int i = 0; i < 8; i++) {
        int p = t + i * 256;               // 2048 = 128 rows x 16 pairs
        int lr = p >> 4, kp = (p & 15) << 1;
        rowp[i] = h + (size_t)s_rows[lr] * H + kp;
        smt[i] = lr >> 4; sm[i] = lr & 15; skp[i] = kp;
    }

    // per-lane base into packed B: column n = n0 + wave*16 + nn, K offset hi*16
    const bf16* bbase = BTUr + (size_t)(n0 + (wave << 4) + nn) * H + (hi << 4);

    // prologue: stage k-step 0
#pragma unroll
    for (int i = 0; i < 8; i++) {
        const float2 f = *(const float2*)(rowp[i]);
        stA(ldsA[0], smt[i], sm[i], skp[i], f.x, f.y);
    }
    __syncthreads();

    for (int ks = 0; ks < 16; ks++) {
        const bf16* cur = ldsA[ks & 1];
        // issue next step's global loads before the WMMA chain
        float2 tmp[8];
        if (ks < 15) {
#pragma unroll
            for (int i = 0; i < 8; i++)
                tmp[i] = *(const float2*)(rowp[i] + (ks + 1) * 32);
            __builtin_prefetch(bbase + (ks + 1) * 32, 0, 1);
        }

        v16bf bf = *(const v16bf*)(bbase + ks * 32);
#pragma unroll
        for (int mt = 0; mt < 8; mt++) {
            v16bf af = *(const v16bf*)(cur + (((mt << 5) + lane) << 4));
            acc[mt] = wmma_bf16(af, bf, acc[mt]);
        }

        if (ks < 15) {
            bf16* nxt = ldsA[(ks + 1) & 1];
#pragma unroll
            for (int i = 0; i < 8; i++)
                stA(nxt, smt[i], sm[i], skp[i], tmp[i].x, tmp[i].y);
        }
        __syncthreads();
    }

    // epilogue: both neighbor reductions are in-lane over the 8 accum slots
    const int col = n0 + (wave << 4) + nn;
    const float bias = Urb[col];
#pragma unroll
    for (int mt = 0; mt < 8; mt++) {
        int e_loc = (mt << 1) + hi;
        float er = embR[(size_t)s_wid[e_loc] * H + col];
        float sh = 0.f, sg = 0.f;
#pragma unroll
        for (int v = 0; v < 8; v++) {
            int hrow = s_rows[(e_loc << 3) + v];
            float hn = h[(size_t)hrow * H + col];
            float r = sigmoidf_(er + acc[mt][v] + bias);
            sh += hn;
            sg += r * hn;
        }
        sum_h [(size_t)(e0 + e_loc) * H + col] = sh;
        sum_gh[(size_t)(e0 + e_loc) * H + col] = sg;
    }
}

// ---------------------------------------------------------------------------
// Dual GEMM: z = sigmoid(embZ[wid] + sum_h @ Wz_bot)
//            p = tanh   (embH[wid] + sum_gh @ Wh_bot)
//            h[out_idx[e]] = (1-z)*sum_h + z*p
// Block: M=64, N=64; 8 waves = 2(M-groups of 2 tiles) x 4(N); 4 WMMA/wave/step.
// ---------------------------------------------------------------------------
__global__ void __launch_bounds__(256)
k_zh_gemm(const float* __restrict__ sum_h, const float* __restrict__ sum_gh,
          const bf16* __restrict__ BTz, const bf16* __restrict__ BTh,
          const float* __restrict__ embZ, const float* __restrict__ embH,
          const int* __restrict__ wid, const int* __restrict__ oidx,
          float* __restrict__ hout) {
    __shared__ __align__(32) bf16 ldsA[2][2 * 4 * 32 * 16];  // 2 buf x {sum_h, sum_gh} x 4 tiles
    __shared__ int s_wid[64];

    const int t  = threadIdx.x;
    const int e0 = blockIdx.x * 64;
    const int n0 = blockIdx.y * 64;
    if (t < 64) s_wid[t] = wid[e0 + t];
    __syncthreads();

    const int wave = t >> 5, lane = t & 31;
    const int mg = wave >> 2, nt = wave & 3;
    const int hi = lane >> 4, nn = lane & 15;
    v8f az[2], ap[2];
#pragma unroll
    for (int m = 0; m < 2; m++)
#pragma unroll
        for (int i = 0; i < 8; i++) { az[m][i] = 0.f; ap[m][i] = 0.f; }

    const bf16* bzb = BTz + (size_t)(n0 + (nt << 4) + nn) * H + (hi << 4);
    const bf16* bpb = BTh + (size_t)(n0 + (nt << 4) + nn) * H + (hi << 4);

    // hoisted staging state: 2048 units = 2 matrices x 64 rows x 16 pairs
    const float* rowp[8];
    int smat[8], smt[8], sm[8], skp[8];
#pragma unroll
    for (int i = 0; i < 8; i++) {
        int u = t + i * 256;
        int mat = u >> 10, r = u & 1023;
        int lr = r >> 4, kp = (r & 15) << 1;
        rowp[i] = (mat ? sum_gh : sum_h) + (size_t)(e0 + lr) * H + kp;
        smat[i] = mat; smt[i] = lr >> 4; sm[i] = lr & 15; skp[i] = kp;
    }

#pragma unroll
    for (int i = 0; i < 8; i++) {
        const float2 f = *(const float2*)(rowp[i]);
        stA(ldsA[0] + smat[i] * 2048, smt[i], sm[i], skp[i], f.x, f.y);
    }
    __syncthreads();

    for (int ks = 0; ks < 16; ks++) {
        const bf16* cur = ldsA[ks & 1];
        float2 tmp[8];
        if (ks < 15) {
#pragma unroll
            for (int i = 0; i < 8; i++)
                tmp[i] = *(const float2*)(rowp[i] + (ks + 1) * 32);
            __builtin_prefetch(bzb + (ks + 1) * 32, 0, 1);
        }

        v16bf bz = *(const v16bf*)(bzb + ks * 32);
        v16bf bp = *(const v16bf*)(bpb + ks * 32);
#pragma unroll
        for (int m = 0; m < 2; m++) {
            int mt = (mg << 1) + m;
            v16bf afz = *(const v16bf*)(cur + (((mt << 5) + lane) << 4));
            v16bf afp = *(const v16bf*)(cur + 2048 + (((mt << 5) + lane) << 4));
            az[m] = wmma_bf16(afz, bz, az[m]);
            ap[m] = wmma_bf16(afp, bp, ap[m]);
        }

        if (ks < 15) {
            bf16* nxt = ldsA[(ks + 1) & 1];
#pragma unroll
            for (int i = 0; i < 8; i++)
                stA(nxt + smat[i] * 2048, smt[i], sm[i], skp[i], tmp[i].x, tmp[i].y);
        }
        __syncthreads();
    }

    const int col = n0 + (nt << 4) + nn;
#pragma unroll
    for (int m = 0; m < 2; m++) {
#pragma unroll
        for (int v = 0; v < 8; v++) {
            int e_loc = (((mg << 1) + m) << 4) + (hi << 3) + v;
            int e = e0 + e_loc;
            float z = sigmoidf_(az[m][v] + embZ[(size_t)s_wid[e_loc] * H + col]);
            float p = tanhf_  (ap[m][v] + embH[(size_t)s_wid[e_loc] * H + col]);
            float sh = sum_h[(size_t)e * H + col];
            int orow = oidx[e];
            hout[(size_t)orow * H + col] = (1.f - z) * sh + z * p;
        }
    }
}

// ---------------------------------------------------------------------------
// Generic GEMM (N=512, K=512), M=64/block: out = act(A @ B + bias [+ embAdd[wid]])
// Used for the 4 emb@W precomputes (relu=0) and the root readout (relu=1).
// ---------------------------------------------------------------------------
__global__ void __launch_bounds__(256)
k_gemm64(const float* __restrict__ A, const bf16* __restrict__ BT,
         const float* __restrict__ bias, const int* __restrict__ wid,
         const float* __restrict__ embAdd, float* __restrict__ out, int relu) {
    __shared__ __align__(32) bf16 ldsA[2][4 * 32 * 16];
    __shared__ int s_wid[64];

    const int t  = threadIdx.x;
    const int e0 = blockIdx.x * 64;
    const int n0 = blockIdx.y * 64;
    if (wid != nullptr && t < 64) s_wid[t] = wid[e0 + t];
    __syncthreads();

    const int wave = t >> 5, lane = t & 31;
    const int mg = wave >> 2, nt = wave & 3;
    const int hi = lane >> 4, nn = lane & 15;
    v8f acc[2];
#pragma unroll
    for (int m = 0; m < 2; m++)
#pragma unroll
        for (int i = 0; i < 8; i++) acc[m][i] = 0.f;

    const bf16* bb = BT + (size_t)(n0 + (nt << 4) + nn) * H + (hi << 4);

    // hoisted staging state: 1024 units = 64 rows x 16 pairs
    const float* rowp[4];
    int smt[4], sm[4], skp[4];
#pragma unroll
    for (int i = 0; i < 4; i++) {
        int u = t + i * 256;
        int lr = u >> 4, kp = (u & 15) << 1;
        rowp[i] = A + (size_t)(e0 + lr) * H + kp;
        smt[i] = lr >> 4; sm[i] = lr & 15; skp[i] = kp;
    }

#pragma unroll
    for (int i = 0; i < 4; i++) {
        const float2 f = *(const float2*)(rowp[i]);
        stA(ldsA[0], smt[i], sm[i], skp[i], f.x, f.y);
    }
    __syncthreads();

    for (int ks = 0; ks < 16; ks++) {
        const bf16* cur = ldsA[ks & 1];
        float2 tmp[4];
        if (ks < 15) {
#pragma unroll
            for (int i = 0; i < 4; i++)
                tmp[i] = *(const float2*)(rowp[i] + (ks + 1) * 32);
        }

        v16bf bf = *(const v16bf*)(bb + ks * 32);
#pragma unroll
        for (int m = 0; m < 2; m++) {
            int mt = (mg << 1) + m;
            v16bf af = *(const v16bf*)(cur + (((mt << 5) + lane) << 4));
            acc[m] = wmma_bf16(af, bf, acc[m]);
        }

        if (ks < 15) {
            bf16* nxt = ldsA[(ks + 1) & 1];
#pragma unroll
            for (int i = 0; i < 4; i++)
                stA(nxt, smt[i], sm[i], skp[i], tmp[i].x, tmp[i].y);
        }
        __syncthreads();
    }

    const int col = n0 + (nt << 4) + nn;
#pragma unroll
    for (int m = 0; m < 2; m++) {
#pragma unroll
        for (int v = 0; v < 8; v++) {
            int e_loc = (((mg << 1) + m) << 4) + (hi << 3) + v;
            int e = e0 + e_loc;
            float val = acc[m][v];
            if (bias)   val += bias[col];
            if (embAdd) val += embAdd[(size_t)s_wid[e_loc] * H + col];
            if (relu)   val = fmaxf(val, 0.f);
            out[(size_t)e * H + col] = val;
        }
    }
}

extern "C" void kernel_launch(void* const* d_in, const int* in_sizes, int n_in,
                              void* d_out, int out_size, void* d_ws, size_t ws_size,
                              hipStream_t stream) {
    (void)in_sizes; (void)n_in; (void)out_size; (void)ws_size;

    const float* emb  = (const float*)d_in[0];
    const float* Wz   = (const float*)d_in[1];   // [2H, H]
    const float* Wz_b = (const float*)d_in[2];
    const float* Wr   = (const float*)d_in[3];   // [H, H]
    const float* Ur   = (const float*)d_in[4];   // [H, H]
    const float* Ur_b = (const float*)d_in[5];
    const float* Wh   = (const float*)d_in[6];   // [2H, H]
    const float* Wh_b = (const float*)d_in[7];
    const float* Ww   = (const float*)d_in[8];   // [2H, H]
    const float* Ww_b = (const float*)d_in[9];
    const int* cur_wid  = (const int*)d_in[10];  // [16, E]
    const int* nei_idx  = (const int*)d_in[11];  // [16, E, 8]
    const int* out_idx  = (const int*)d_in[12];  // [16, E]
    const int* root_wid = (const int*)d_in[13];  // [B]
    const int* root_nei = (const int*)d_in[14];  // [B, 8]

    float* hbuf     = (float*)d_out;                   // (N_MSG+1) x H
    float* root_out = hbuf + (size_t)(N_MSG + 1) * H;  // B x H

    float* ws     = (float*)d_ws;
    float* embZ   = ws;                                // VOCAB x H
    float* embR   = embZ + (size_t)VOCAB * H;
    float* embH   = embR + (size_t)VOCAB * H;
    float* embW   = embH + (size_t)VOCAB * H;
    float* sum_h  = embW + (size_t)VOCAB * H;          // E x H
    float* sum_gh = sum_h + (size_t)E * H;             // E x H

    bf16* bt      = (bf16*)(sum_gh + (size_t)E * H);   // 8 packed bf16 [512][512]
    bf16* BT_Ur   = bt;
    bf16* BT_Wz_t = bt + 1 * (size_t)H * H;
    bf16* BT_Wz_b = bt + 2 * (size_t)H * H;
    bf16* BT_Wr   = bt + 3 * (size_t)H * H;
    bf16* BT_Wh_t = bt + 4 * (size_t)H * H;
    bf16* BT_Wh_b = bt + 5 * (size_t)H * H;
    bf16* BT_Ww_t = bt + 6 * (size_t)H * H;
    bf16* BT_Ww_b = bt + 7 * (size_t)H * H;

    // h starts all-zero; PAD row (index N_MSG) stays zero forever.
    hipMemsetAsync(hbuf, 0, (size_t)(N_MSG + 1) * H * sizeof(float), stream);

    const dim3 blk(256);
    const int packGrid = (H * H / 2) / 256;

    // Pack all B operands to bf16-transposed once.
    k_pack_bt<<<packGrid, blk, 0, stream>>>(Ur, BT_Ur);
    k_pack_bt<<<packGrid, blk, 0, stream>>>(Wz, BT_Wz_t);
    k_pack_bt<<<packGrid, blk, 0, stream>>>(Wz + (size_t)H * H, BT_Wz_b);
    k_pack_bt<<<packGrid, blk, 0, stream>>>(Wr, BT_Wr);
    k_pack_bt<<<packGrid, blk, 0, stream>>>(Wh, BT_Wh_t);
    k_pack_bt<<<packGrid, blk, 0, stream>>>(Wh + (size_t)H * H, BT_Wh_b);
    k_pack_bt<<<packGrid, blk, 0, stream>>>(Ww, BT_Ww_t);
    k_pack_bt<<<packGrid, blk, 0, stream>>>(Ww + (size_t)H * H, BT_Ww_b);

    // Precompute embedding projections (folds every x@W GEMM into a gather).
    k_gemm64<<<dim3(VOCAB / 64, 8), blk, 0, stream>>>(emb, BT_Wz_t, Wz_b, nullptr, nullptr, embZ, 0);
    k_gemm64<<<dim3(VOCAB / 64, 8), blk, 0, stream>>>(emb, BT_Wr,  nullptr, nullptr, nullptr, embR, 0);
    k_gemm64<<<dim3(VOCAB / 64, 8), blk, 0, stream>>>(emb, BT_Wh_t, Wh_b, nullptr, nullptr, embH, 0);
    k_gemm64<<<dim3(VOCAB / 64, 8), blk, 0, stream>>>(emb, BT_Ww_t, Ww_b, nullptr, nullptr, embW, 0);

    for (int it = 0; it < N_ITERS; it++) {
        const int* nei_it  = nei_idx + (size_t)it * E * MAX_NB;
        const int* wid_it  = cur_wid + (size_t)it * E;
        const int* oidx_it = out_idx + (size_t)it * E;

        k_r2_gemm<<<dim3(E / 16, 4), blk, 0, stream>>>(hbuf, nei_it, wid_it, embR,
                                                       BT_Ur, Ur_b, sum_h, sum_gh);
        k_zh_gemm<<<dim3(E / 64, 8), blk, 0, stream>>>(sum_h, sum_gh, BT_Wz_b, BT_Wh_b,
                                                       embZ, embH, wid_it, oidx_it, hbuf);
    }

    // Root readout: relu(embW[root_wid] + (sum_j h[root_nei]) @ Ww_bot)
    k_gather_sum<<<(B_ROOT * 128) / 256, blk, 0, stream>>>(hbuf, root_nei, sum_h, B_ROOT);
    k_gemm64<<<dim3(B_ROOT / 64, 8), blk, 0, stream>>>(sum_h, BT_Ww_b, nullptr, root_wid,
                                                       embW, root_out, 1);
}
